// MultiHeadSelfAttention_14285061226765
// MI455X (gfx1250) — matrix operand order repaired and checked by reading, not compiled, not run on hardware
//
#include <hip/hip_runtime.h>

// ---------------------------------------------------------------------------
// MHSA forward for MI455X (gfx1250): bf16 WMMA everywhere, f32 accumulate.
// B=4, S=2048, E=1024, H=16, D=64.
// ---------------------------------------------------------------------------

#define B_DIM 4
#define S_DIM 2048
#define E_DIM 1024
#define H_DIM 16
#define D_DIM 64

// 1/sqrt(D) * log2(e): folded into the Q projection so the softmax runs in
// base-2 with zero pre-scaling (v_exp_f32 is 2^x on AMD).
#define Q_SCALE 0.18033688011112042f

#if __has_builtin(__builtin_amdgcn_exp2f)
#define EXP2F(x) __builtin_amdgcn_exp2f(x)
#else
#define EXP2F(x) exp2f(x)
#endif

typedef __bf16 bf16_t;
typedef __attribute__((ext_vector_type(4)))  bf16_t v4bf;
typedef __attribute__((ext_vector_type(8)))  bf16_t v8bf;
typedef __attribute__((ext_vector_type(16))) bf16_t v16bf;
typedef __attribute__((ext_vector_type(8)))  float  v8f;

__device__ __forceinline__ v8f wmma_bf16(v16bf a, v16bf b, v8f c) {
  // v_wmma_f32_16x16x32_bf16: (neg_a, A, neg_b, B, c_mod, C, reuse_a, reuse_b)
  return __builtin_amdgcn_wmma_f32_16x16x32_bf16(false, a, false, b, (short)0, c,
                                                 false, false);
}

// --- CDNA5 async copy: global -> LDS, 16B per lane, tracked by ASYNCcnt. ----
__device__ __forceinline__ void async_copy_b128(unsigned lds_addr,
                                                const void* gptr) {
  asm volatile("global_load_async_to_lds_b128 %0, %1, off"
               :: "v"(lds_addr), "v"(gptr)
               : "memory");
}
__device__ __forceinline__ void wait_async0() {
  asm volatile("s_wait_asynccnt 0x0" ::: "memory");
}
// Wave-local LDS ordering fence (DS ops are in-order within a wave).
__device__ __forceinline__ void lds_fence_wave() {
  asm volatile("s_wait_dscnt 0x0" ::: "memory");
}

// --- DPP16 butterfly reductions over each 16-lane group (pure VALU). --------
template <int CTRL>
__device__ __forceinline__ float dpp_swap(float x) {
  return __int_as_float(__builtin_amdgcn_update_dpp(
      0, __float_as_int(x), CTRL, 0xF, 0xF, true));
}
__device__ __forceinline__ float red16_max(float x) {
  x = fmaxf(x, dpp_swap<0xB1>(x));   // quad_perm(1,0,3,2): xor 1
  x = fmaxf(x, dpp_swap<0x4E>(x));   // quad_perm(2,3,0,1): xor 2
  x = fmaxf(x, dpp_swap<0x141>(x));  // row_half_mirror:    xor 7
  x = fmaxf(x, dpp_swap<0x140>(x));  // row_mirror:         xor 15
  return x;
}
__device__ __forceinline__ float red16_sum(float x) {
  x += dpp_swap<0xB1>(x);
  x += dpp_swap<0x4E>(x);
  x += dpp_swap<0x141>(x);
  x += dpp_swap<0x140>(x);
  return x;
}

// A-fragment (16x32 bf16, MxK), source row-major with leading dim ld.
// Layout (ISA 7.12.2): lane = M, VGPR0-3 = K {0..7}+8*halfwave,
// VGPR4-7 = K {16..23}+8*halfwave  ->  f[j] = row[ half*8 + (j<8 ? j : j+8) ].
__device__ __forceinline__ v16bf frag_a(const bf16_t* tile, int ld) {
  const int lane = threadIdx.x & 31;
  const int m = lane & 15, hl = lane >> 4;
  const bf16_t* rp = tile + (size_t)m * ld + hl * 8;
  v8bf lo = *(const v8bf*)(rp);
  v8bf hi = *(const v8bf*)(rp + 16);
  return __builtin_shufflevector(lo, hi, 0, 1, 2, 3, 4, 5, 6, 7,
                                 8, 9, 10, 11, 12, 13, 14, 15);
}

// B-fragment (32x16 bf16, KxN), sourced from a row-major NxK ("transposed")
// matrix with leading dim ld: lane = N, same K order as frag_a
// ->  f[j] = rowN[ half*8 + (j<8 ? j : j+8) ].
__device__ __forceinline__ v16bf frag_b(const bf16_t* tileT, int ld) {
  const int lane = threadIdx.x & 31;
  const int n = lane & 15, hl = lane >> 4;
  const bf16_t* rp = tileT + (size_t)n * ld + hl * 8;
  v8bf lo = *(const v8bf*)(rp);
  v8bf hi = *(const v8bf*)(rp + 16);
  return __builtin_shufflevector(lo, hi, 0, 1, 2, 3, 4, 5, 6, 7,
                                 8, 9, 10, 11, 12, 13, 14, 15);
}

// ---------------------------------------------------------------------------
// Elementwise f32 -> bf16 (4 elems/thread).
// ---------------------------------------------------------------------------
__global__ __launch_bounds__(256) void f32_to_bf16_kernel(
    const float* __restrict__ src, bf16_t* __restrict__ dst, int n) {
  int i = (blockIdx.x * blockDim.x + threadIdx.x) * 4;
  if (i + 3 < n) {
    const float4 f = *(const float4*)(src + i);
    v4bf o = {(bf16_t)f.x, (bf16_t)f.y, (bf16_t)f.z, (bf16_t)f.w};
    *(v4bf*)(dst + i) = o;
  } else {
    for (int j = i; j < n; ++j) dst[j] = (bf16_t)src[j];
  }
}

// ---------------------------------------------------------------------------
// Transpose-convert weight: Wt[n][k] = (bf16) W[k][n]   (ExE).
// ---------------------------------------------------------------------------
__global__ __launch_bounds__(256) void transpose_to_bf16_kernel(
    const float* __restrict__ W, bf16_t* __restrict__ Wt) {
  __shared__ float tile[32][33];
  const int tx = threadIdx.x, ty = threadIdx.y;
  const int x = blockIdx.x * 32 + tx;
  const int y0 = blockIdx.y * 32;
#pragma unroll
  for (int i = ty; i < 32; i += 8) tile[i][tx] = W[(size_t)(y0 + i) * E_DIM + x];
  __syncthreads();
  const int xo = y0 + tx;
  const int n0 = blockIdx.x * 32;
#pragma unroll
  for (int i = ty; i < 32; i += 8)
    Wt[(size_t)(n0 + i) * E_DIM + xo] = (bf16_t)tile[tx][i];
}

// ---------------------------------------------------------------------------
// Tiled WMMA GEMM: C[M,N] = (A[M,K] * Bt[N,K]^T + bias[N]) * oscale.
// Block tile 128x128, BK=32, 8 waves (4 over M x 2 over N), each wave 2x4
// 16x16 accumulators. Double-buffered async-to-LDS staging (ASYNCcnt).
// MODE 0: bf16 out, head-split [B,H,S,D].  MODE 1: bf16 out, [B,H,D,S].
// MODE 2: f32 out, plain [M,E].
// ---------------------------------------------------------------------------
template <int MODE>
__global__ __launch_bounds__(256) void gemm_bf16_kernel(
    const bf16_t* __restrict__ A, const bf16_t* __restrict__ Bt,
    const float* __restrict__ bias, void* __restrict__ out, float oscale) {
  __shared__ __align__(16) bf16_t As[2][128 * 40];
  __shared__ __align__(16) bf16_t Bs[2][128 * 40];

  const int tid = threadIdx.x;
  const int wid = tid >> 5, lane = tid & 31;
  const int bm = blockIdx.y * 128, bn = blockIdx.x * 128;
  const int wm = (wid >> 1) * 32;  // 4 waves over M (32 rows each)
  const int wn = (wid & 1) * 64;   // 2 waves over N (64 cols each)

  v8f acc[2][4];
#pragma unroll
  for (int i = 0; i < 2; ++i)
#pragma unroll
    for (int j = 0; j < 4; ++j) acc[i][j] = (v8f){0, 0, 0, 0, 0, 0, 0, 0};

  const int lrow = tid >> 1;        // 0..127
  const int lcol = (tid & 1) * 16;  // 0 or 16

  const bf16_t* ga = A + (size_t)(bm + lrow) * E_DIM + lcol;
  const bf16_t* gb = Bt + (size_t)(bn + lrow) * E_DIM + lcol;

  // LDS byte addresses for this thread's 32B staging slot in each buffer
  // (low 32 bits of a flat shared pointer == LDS byte address).
  unsigned laA[2], laB[2];
  laA[0] = (unsigned)(uintptr_t)&As[0][lrow * 40 + lcol];
  laA[1] = (unsigned)(uintptr_t)&As[1][lrow * 40 + lcol];
  laB[0] = (unsigned)(uintptr_t)&Bs[0][lrow * 40 + lcol];
  laB[1] = (unsigned)(uintptr_t)&Bs[1][lrow * 40 + lcol];

  // Prologue: stage K-tile 0 into buffer 0.
  async_copy_b128(laA[0], ga);
  async_copy_b128(laA[0] + 16, ga + 8);
  async_copy_b128(laB[0], gb);
  async_copy_b128(laB[0] + 16, gb + 8);

  int cur = 0;
  for (int k0 = 0; k0 < E_DIM; k0 += 32) {
    wait_async0();     // this wave's async writes have landed
    __syncthreads();   // publish: all waves' writes visible

    if (k0 + 32 < E_DIM) {  // kick off next tile into the other buffer
      const int nxt = cur ^ 1;
      const bf16_t* na = ga + k0 + 32;
      const bf16_t* nb = gb + k0 + 32;
      async_copy_b128(laA[nxt], na);
      async_copy_b128(laA[nxt] + 16, na + 8);
      async_copy_b128(laB[nxt], nb);
      async_copy_b128(laB[nxt] + 16, nb + 8);
    }

    const bf16_t* Ac = As[cur];
    const bf16_t* Bc = Bs[cur];
    v16bf a0 = frag_a(Ac + (wm + 0) * 40, 40);
    v16bf a1 = frag_a(Ac + (wm + 16) * 40, 40);
    v16bf b0 = frag_b(Bc + (wn + 0) * 40, 40);
    v16bf b1 = frag_b(Bc + (wn + 16) * 40, 40);
    v16bf b2 = frag_b(Bc + (wn + 32) * 40, 40);
    v16bf b3 = frag_b(Bc + (wn + 48) * 40, 40);

    acc[0][0] = wmma_bf16(a0, b0, acc[0][0]);
    acc[0][1] = wmma_bf16(a0, b1, acc[0][1]);
    acc[0][2] = wmma_bf16(a0, b2, acc[0][2]);
    acc[0][3] = wmma_bf16(a0, b3, acc[0][3]);
    acc[1][0] = wmma_bf16(a1, b0, acc[1][0]);
    acc[1][1] = wmma_bf16(a1, b1, acc[1][1]);
    acc[1][2] = wmma_bf16(a1, b2, acc[1][2]);
    acc[1][3] = wmma_bf16(a1, b3, acc[1][3]);

    __syncthreads();   // all waves done reading buffer `cur`
    cur ^= 1;
  }

  // Epilogue. C layout: VGPR r -> row r + 8*halfwave, col = lane&15.
  const int hl = lane >> 4;
  const int nloc = lane & 15;
#pragma unroll
  for (int mt = 0; mt < 2; ++mt) {
#pragma unroll
    for (int nt = 0; nt < 4; ++nt) {
      const int ng = bn + wn + nt * 16 + nloc;
      const float bv = bias[ng];
#pragma unroll
      for (int r = 0; r < 8; ++r) {
        const int mg = bm + wm + mt * 16 + r + 8 * hl;
        const float val = (acc[mt][nt][r] + bv) * oscale;
        if constexpr (MODE == 2) {
          ((float*)out)[(size_t)mg * E_DIM + ng] = val;
        } else {
          const int bb = mg >> 11, ss = mg & (S_DIM - 1);
          const int hh = ng >> 6, dd = ng & (D_DIM - 1);
          if constexpr (MODE == 0) {  // [B,H,S,D]
            ((bf16_t*)out)[(((size_t)bb * H_DIM + hh) * S_DIM + ss) * D_DIM + dd] =
                (bf16_t)val;
          } else {                    // [B,H,D,S]
            ((bf16_t*)out)[(((size_t)bb * H_DIM + hh) * D_DIM + dd) * S_DIM + ss] =
                (bf16_t)val;
          }
        }
      }
    }
  }
}

// ---------------------------------------------------------------------------
// Flash attention: Q[B,H,S,D] (pre-scaled by 1/sqrt(D)*log2e), K[B,H,S,D],
// Vt[B,H,D,S] (all bf16) -> attn[B,S,E] bf16. One wave per 16-row query tile;
// online base-2 softmax over 32-key tiles; scores and P*V on
// v_wmma_f32_16x16x32_bf16. Row stats via DPP16 butterflies; P staged through
// wave-private LDS (wave-local s_wait_dscnt fence, no inter-wave barriers).
// ---------------------------------------------------------------------------
__global__ __launch_bounds__(256) void attention_kernel(
    const bf16_t* __restrict__ Q, const bf16_t* __restrict__ K,
    const bf16_t* __restrict__ Vt, bf16_t* __restrict__ attn) {
  __shared__ __align__(16) bf16_t Plds[8 * 16 * 32];  // per-wave 16x32 P tile

  const int tid = threadIdx.x, wid = tid >> 5, lane = tid & 31;
  const int nloc = lane & 15, hl = lane >> 4;
  const int bh = blockIdx.y;  // b*H + h
  const int b = bh >> 4, h = bh & (H_DIM - 1);
  const int q0 = blockIdx.x * 128 + wid * 16;

  const bf16_t* Qb = Q + (size_t)bh * S_DIM * D_DIM;
  const bf16_t* Kb = K + (size_t)bh * S_DIM * D_DIM;
  const bf16_t* Vb = Vt + (size_t)bh * D_DIM * S_DIM;
  bf16_t* Pw = Plds + wid * 16 * 32;

  // Q fragments for this wave's 16 rows (k = 0..31 and 32..63), kept in regs.
  const v16bf qa0 = frag_a(Qb + (size_t)q0 * D_DIM, D_DIM);
  const v16bf qa1 = frag_a(Qb + (size_t)q0 * D_DIM + 32, D_DIM);

  v8f o[4];
#pragma unroll
  for (int nt = 0; nt < 4; ++nt) o[nt] = (v8f){0, 0, 0, 0, 0, 0, 0, 0};
  float mrow[8], lsum[8];
#pragma unroll
  for (int r = 0; r < 8; ++r) { mrow[r] = -1e30f; lsum[r] = 0.0f; }

  const v8f zero = (v8f){0, 0, 0, 0, 0, 0, 0, 0};

  for (int kt = 0; kt < S_DIM; kt += 32) {
    // K^T fragments: B(k=d, n=key) -> per-lane contiguous reads of K rows.
    v16bf kb00 = frag_b(Kb + (size_t)kt * D_DIM, D_DIM);
    v16bf kb01 = frag_b(Kb + (size_t)kt * D_DIM + 32, D_DIM);
    v16bf kb10 = frag_b(Kb + (size_t)(kt + 16) * D_DIM, D_DIM);
    v16bf kb11 = frag_b(Kb + (size_t)(kt + 16) * D_DIM + 32, D_DIM);

    v8f s0 = wmma_bf16(qa0, kb00, zero);
    s0 = wmma_bf16(qa1, kb01, s0);
    v8f s1 = wmma_bf16(qa0, kb10, zero);
    s1 = wmma_bf16(qa1, kb11, s1);

    // Online base-2 softmax; row stats reduced over 16-lane groups via DPP.
    float p0[8], p1[8];
#pragma unroll
    for (int r = 0; r < 8; ++r) {
      const float x0 = s0[r];   // already scaled by 1/sqrt(D)*log2e
      const float x1 = s1[r];
      const float mv = red16_max(fmaxf(x0, x1));
      const float mnew = fmaxf(mrow[r], mv);
      const float alpha = EXP2F(mrow[r] - mnew);
      p0[r] = EXP2F(x0 - mnew);
      p1[r] = EXP2F(x1 - mnew);
      const float ps = red16_sum(p0[r] + p1[r]);
      lsum[r] = lsum[r] * alpha + ps;
      mrow[r] = mnew;
#pragma unroll
      for (int nt = 0; nt < 4; ++nt) o[nt][r] *= alpha;
    }

    // C-layout P -> wave-private LDS (row-major 16x32) -> A fragment.
#pragma unroll
    for (int r = 0; r < 8; ++r) {
      Pw[(r + 8 * hl) * 32 + nloc] = (bf16_t)p0[r];
      Pw[(r + 8 * hl) * 32 + 16 + nloc] = (bf16_t)p1[r];
    }
    lds_fence_wave();  // DS in-order per wave; fence before re-reading
    v16bf pa = frag_a(Pw, 32);

    // P (16x32) * V (32x64): B(k=key, n=d) from Vt rows (contiguous in keys).
#pragma unroll
    for (int nt = 0; nt < 4; ++nt) {
      v16bf vb = frag_b(Vb + (size_t)(nt * 16) * S_DIM + kt, S_DIM);
      o[nt] = wmma_bf16(pa, vb, o[nt]);
    }
  }

  // Normalize and store merged heads: attn[b][s][h*64+d].
#pragma unroll
  for (int r = 0; r < 8; ++r) {
    const float inv = 1.0f / lsum[r];
    const int s = q0 + r + 8 * hl;
#pragma unroll
    for (int nt = 0; nt < 4; ++nt) {
      const int d = nt * 16 + nloc;
      attn[((size_t)b * S_DIM + s) * E_DIM + h * D_DIM + d] =
          (bf16_t)(o[nt][r] * inv);
    }
  }
}

// ---------------------------------------------------------------------------
// Launch. Workspace layout (needs 88 MB of d_ws):
//   [0,16M)   Xb bf16 [B*S,E]   [16..24M) Wq/Wk/Wv/Wo^T bf16 [E,E]
//   [24,40M)  Q [B,H,S,D]  [40,56M) K [B,H,S,D]  [56,72M) Vt [B,H,D,S]
//   [72,88M)  attn bf16 [B,S,E]
// ---------------------------------------------------------------------------
extern "C" void kernel_launch(void* const* d_in, const int* in_sizes, int n_in,
                              void* d_out, int out_size, void* d_ws, size_t ws_size,
                              hipStream_t stream) {
  (void)in_sizes; (void)n_in; (void)out_size; (void)ws_size;
  const float* X  = (const float*)d_in[0];
  const float* Wq = (const float*)d_in[1];
  const float* bq = (const float*)d_in[2];
  const float* Wk = (const float*)d_in[3];
  const float* bk = (const float*)d_in[4];
  const float* Wv = (const float*)d_in[5];
  const float* bv = (const float*)d_in[6];
  const float* Wo = (const float*)d_in[7];
  const float* bo = (const float*)d_in[8];

  char* ws = (char*)d_ws;
  const size_t MB = (size_t)1 << 20;
  bf16_t* Xb  = (bf16_t*)(ws + 0 * MB);
  bf16_t* Wqb = (bf16_t*)(ws + 16 * MB);
  bf16_t* Wkb = (bf16_t*)(ws + 18 * MB);
  bf16_t* Wvb = (bf16_t*)(ws + 20 * MB);
  bf16_t* Wob = (bf16_t*)(ws + 22 * MB);
  bf16_t* Qh  = (bf16_t*)(ws + 24 * MB);
  bf16_t* Kh  = (bf16_t*)(ws + 40 * MB);
  bf16_t* Vth = (bf16_t*)(ws + 56 * MB);
  bf16_t* Ah  = (bf16_t*)(ws + 72 * MB);

  const int n_x = B_DIM * S_DIM * E_DIM;  // 8388608
  f32_to_bf16_kernel<<<n_x / (256 * 4), 256, 0, stream>>>(X, Xb, n_x);

  dim3 tb(32, 8), tg(E_DIM / 32, E_DIM / 32);
  transpose_to_bf16_kernel<<<tg, tb, 0, stream>>>(Wq, Wqb);
  transpose_to_bf16_kernel<<<tg, tb, 0, stream>>>(Wk, Wkb);
  transpose_to_bf16_kernel<<<tg, tb, 0, stream>>>(Wv, Wvb);
  transpose_to_bf16_kernel<<<tg, tb, 0, stream>>>(Wo, Wob);

  dim3 gg(E_DIM / 128, (B_DIM * S_DIM) / 128);  // (8, 64)
  gemm_bf16_kernel<0><<<gg, 256, 0, stream>>>(Xb, Wqb, bq, Qh, Q_SCALE);
  gemm_bf16_kernel<0><<<gg, 256, 0, stream>>>(Xb, Wkb, bk, Kh, 1.0f);
  gemm_bf16_kernel<1><<<gg, 256, 0, stream>>>(Xb, Wvb, bv, Vth, 1.0f);

  attention_kernel<<<dim3(S_DIM / 128, B_DIM * H_DIM), 256, 0, stream>>>(Qh, Kh, Vth, Ah);

  gemm_bf16_kernel<2><<<gg, 256, 0, stream>>>(Ah, Wob, bo, d_out, 1.0f);
}
